// QMatMul_30786325577996
// MI455X (gfx1250) — 8-bit matrix operand order repaired and checked by reading, compiled, not run on hardware
//
#include <hip/hip_runtime.h>
#include <cstdint>


typedef __attribute__((ext_vector_type(8))) int v8i;
typedef __attribute__((ext_vector_type(4))) int v4i;

union Frag8 {
  v8i v;
  v4i q[2];
  unsigned long long u[4];
  int i[8];
};

__device__ __forceinline__ void async_b128(uint32_t lds_off, const void* gptr) {
  asm volatile("global_load_async_to_lds_b128 %0, %1, off"
               :: "v"(lds_off), "v"(gptr) : "memory");
}
__device__ __forceinline__ void wait_async0() {
  asm volatile("s_wait_asynccnt 0" ::: "memory");
}

__device__ __forceinline__ signed char fq_s8(float x, float inv_s, float z, int zi) {
  int q = (int)rintf(fmaf(x, inv_s, z));
  q = q < 0 ? 0 : (q > 255 ? 255 : q);
  return (signed char)(q - zi);
}

__global__ __launch_bounds__(256)
void quantize_s8(const float* __restrict__ x, signed char* __restrict__ q,
                 const float* __restrict__ sp, const float* __restrict__ zp,
                 long n) {
  const float s = sp[0], z = zp[0];
  const float inv = 1.0f / s;
  const int zi = (int)rintf(z);
  long i0 = ((long)blockIdx.x * 256 + threadIdx.x) * 16;
  if (i0 >= n) return;
  union { signed char c[16]; int4 v; } out;
  const float4* xp = (const float4*)(x + i0);
#pragma unroll
  for (int j = 0; j < 4; ++j) {
    float4 f = xp[j];
    out.c[j * 4 + 0] = fq_s8(f.x, inv, z, zi);
    out.c[j * 4 + 1] = fq_s8(f.y, inv, z, zi);
    out.c[j * 4 + 2] = fq_s8(f.z, inv, z, zi);
    out.c[j * 4 + 3] = fq_s8(f.w, inv, z, zi);
  }
  *(int4*)(q + i0) = out.v;
}

__global__ __launch_bounds__(256)
void quantize_s8_T(const float* __restrict__ x, signed char* __restrict__ qT,
                   const float* __restrict__ sp, const float* __restrict__ zp,
                   int K, int N) {
  __shared__ signed char tile[32][36];
  const float s = sp[0], z = zp[0];
  const float inv = 1.0f / s;
  const int zi = (int)rintf(z);
  const int b = blockIdx.z;
  const int k0 = blockIdx.y * 32;
  const int n0 = blockIdx.x * 32;
  const int tid = threadIdx.x;
  {
    int kr = tid >> 3;
    int nq = (tid & 7) * 4;
    float4 f = *(const float4*)(x + ((size_t)b * K + k0 + kr) * N + n0 + nq);
    tile[kr][nq + 0] = fq_s8(f.x, inv, z, zi);
    tile[kr][nq + 1] = fq_s8(f.y, inv, z, zi);
    tile[kr][nq + 2] = fq_s8(f.z, inv, z, zi);
    tile[kr][nq + 3] = fq_s8(f.w, inv, z, zi);
  }
  __syncthreads();
  {
    int nn = tid >> 3;
    int kq = (tid & 7) * 4;
    union { signed char c[4]; int v; } o;
#pragma unroll
    for (int e = 0; e < 4; ++e) o.c[e] = tile[kq + e][nn];
    *(int*)(qT + ((size_t)b * N + n0 + nn) * K + k0 + kq) = o.v;
  }
}

__global__ __launch_bounds__(256)
void qgemm_i8_wmma(const signed char* __restrict__ Aq,
                   const signed char* __restrict__ BqT,
                   float* __restrict__ C,
                   const float* __restrict__ s1p, const float* __restrict__ s2p,
                   int M, int N, int K) {
  constexpr int LDA = 80, LDB = 80, BK = 64;
  __shared__ __align__(16) signed char sA[2][128 * LDA];
  __shared__ __align__(16) signed char sB[2][128 * LDB];

  const int b  = blockIdx.z;
  const int n0 = blockIdx.x * 128;
  const int m0 = blockIdx.y * 128;
  const int tid   = threadIdx.x;
  const int lane  = tid & 31;
  const int wid   = tid >> 5;
  const int wm    = wid >> 1;
  const int wn    = wid & 1;
  const int mlane = lane & 15;
  const int khalf = lane >> 4;

  const signed char* Ab = Aq  + ((size_t)b * M + m0) * K;
  const signed char* Bb = BqT + ((size_t)b * N + n0) * K;

  auto stage = [&](int buf, int kt) {
    const signed char* Ak = Ab + kt * BK;
    const signed char* Bk = Bb + kt * BK;
#pragma unroll
    for (int it = 0; it < 2; ++it) {
      int o   = (tid + it * 256) * 16;
      int row = o >> 6;
      int col = o & 63;
      async_b128((uint32_t)(uintptr_t)&sA[buf][row * LDA + col],
                 Ak + (size_t)row * K + col);
      async_b128((uint32_t)(uintptr_t)&sB[buf][row * LDB + col],
                 Bk + (size_t)row * K + col);
    }
  };

  const v8i zero = {0, 0, 0, 0, 0, 0, 0, 0};
  Frag8 acc[2][4];
#pragma unroll
  for (int mi = 0; mi < 2; ++mi)
#pragma unroll
    for (int ni = 0; ni < 4; ++ni) acc[mi][ni].v = zero;

  const int nk = K / BK;
  stage(0, 0);
  wait_async0();
  __syncthreads();

  for (int kt = 0; kt < nk; ++kt) {
    const int cur = kt & 1;
    if (kt + 1 < nk) stage(cur ^ 1, kt + 1);

    const signed char* As = sA[cur];
    const signed char* Bs = sB[cur];

    Frag8 af[2];
#pragma unroll
    for (int mi = 0; mi < 2; ++mi) {
      const signed char* p = As + (wm * 32 + mi * 16 + mlane) * LDA + khalf * 16;
      af[mi].q[0] = *(const v4i*)(p);
      af[mi].q[1] = *(const v4i*)(p + 32);
    }
    Frag8 bf[4];
#pragma unroll
    for (int ni = 0; ni < 4; ++ni) {
      const signed char* p = Bs + (wn * 64 + ni * 16 + mlane) * LDB + khalf * 16;
      bf[ni].q[0] = *(const v4i*)(p);
      bf[ni].q[1] = *(const v4i*)(p + 32);
    }

#pragma unroll
    for (int mi = 0; mi < 2; ++mi)
#pragma unroll
      for (int ni = 0; ni < 4; ++ni)
        acc[mi][ni].v = __builtin_amdgcn_wmma_i32_16x16x64_iu8(
             true, af[mi].v,  true, bf[ni].v,
            acc[mi][ni].v,  false,  false);

    wait_async0();
    __syncthreads();
  }

  const float s12 = s1p[0] * s2p[0];
  float* Cb = C + (size_t)b * M * N;
#pragma unroll
  for (int mi = 0; mi < 2; ++mi) {
#pragma unroll
    for (int ni = 0; ni < 4; ++ni) {
      int mg = m0 + wm * 32 + mi * 16 + khalf * 8;
      int ng = n0 + wn * 64 + ni * 16 + mlane;
      float* cp = Cb + (size_t)mg * N + ng;
#pragma unroll
      for (int r = 0; r < 8; ++r)
        cp[(size_t)r * N] = (float)acc[mi][ni].i[r] * s12;
    }
  }
}

extern "C" void kernel_launch(void* const* d_in, const int* in_sizes, int n_in,
                              void* d_out, int out_size, void* d_ws, size_t ws_size,
                              hipStream_t stream) {
  constexpr int B = 4, M = 2048, K = 4096, N = 4096;
  const float* x1 = (const float*)d_in[0];
  const float* x2 = (const float*)d_in[1];
  const float* s1 = (const float*)d_in[2];
  const float* z1 = (const float*)d_in[3];
  const float* s2 = (const float*)d_in[4];
  const float* z2 = (const float*)d_in[5];
  float* out = (float*)d_out;

  signed char* aq = (signed char*)d_ws;
  signed char* bT = aq + (size_t)B * M * K;

  const long n1 = (long)B * M * K;
  quantize_s8<<<(int)(n1 / (256L * 16)), 256, 0, stream>>>(x1, aq, s1, z1, n1);
  quantize_s8_T<<<dim3(N / 32, K / 32, B), 256, 0, stream>>>(x2, bT, s2, z2, K, N);
  qgemm_i8_wmma<<<dim3(N / 128, M / 128, B), 256, 0, stream>>>(aq, bT, out,
                                                               s1, s2, M, N, K);
}
